// AdvancedFastSpeech2_76991583748685
// MI455X (gfx1250) — matrix operand order repaired and checked by reading, not compiled, not run on hardware
//
#include <hip/hip_runtime.h>
#include <cstdint>
#include <cstddef>

// ---------------- model dimensions ----------------
#define BZ   8
#define TT   256
#define HH   256
#define MELX 1536
#define NMEL 80
#define BT   (BZ*TT)

typedef __attribute__((ext_vector_type(16))) _Float16 v16h;
typedef __attribute__((ext_vector_type(8)))  float    v8f;
typedef __fp16 __attribute__((ext_vector_type(2)))    v2fp;  // cvt_pkrtz return type

union AFrag { v16h v; uint4 q[2]; _Float16 e[16]; };
union CFrag { v8f  v; float f[8]; };
union P2    { v2fp h; _Float16 f[2]; unsigned int u; };

__device__ inline v8f wmma32(v16h a, v16h b, v8f c) {
  return __builtin_amdgcn_wmma_f32_16x16x32_f16(false, a, false, b, (short)0, c,
                                                false, false);
}

// A-fragment (16x32 f16): lane m=lane&15, h=lane>>4; elems 0..7 = K h*8+0..7,
// elems 8..15 = K 16+h*8+0..7 -> two ds_load_b128. ld must be multiple of 8.
__device__ inline v16h ld_fragA(const _Float16* t, int ld) {
  const int lane = threadIdx.x & 31;
  const int m = lane & 15, h = (lane >> 4) & 1;
  AFrag f;
  const _Float16* p = t + m * ld + h * 8;
  f.q[0] = *(const uint4*)(p);
  f.q[1] = *(const uint4*)(p + 16);
  return f.v;
}

// B-fragment (32x16 f16): lane n=lane&15, h=lane>>4; elems 0..7 = K h*8+0..7, elems 8..15 = K 16+h*8+0..7.
// LDS tile layout [n][k] (weight row-major [out][in]).
__device__ inline v16h ld_fragB(const _Float16* t, int ld) {
  const int lane = threadIdx.x & 31;
  const int n = lane & 15, h = (lane >> 4) & 1;
  AFrag f;
  const _Float16* p = t + n * ld + h * 8;
  f.q[0] = *(const uint4*)(p);
  f.q[1] = *(const uint4*)(p + 16);
  return f.v;
}

// packed f32->f16 conversion: 4x v_cvt_pk_rtz_f16_f32 per 8 elements
__device__ inline uint4 cvt8(float4 f0, float4 f1, float scale) {
  P2 p0, p1, p2, p3;
  p0.h = __builtin_amdgcn_cvt_pkrtz(f0.x * scale, f0.y * scale);
  p1.h = __builtin_amdgcn_cvt_pkrtz(f0.z * scale, f0.w * scale);
  p2.h = __builtin_amdgcn_cvt_pkrtz(f1.x * scale, f1.y * scale);
  p3.h = __builtin_amdgcn_cvt_pkrtz(f1.z * scale, f1.w * scale);
  uint4 u;
  u.x = p0.u; u.y = p1.u; u.z = p2.u; u.w = p3.u;
  return u;
}

__device__ inline _Float16 cvt1(float x) {
  P2 p;
  p.h = __builtin_amdgcn_cvt_pkrtz(x, x);
  return p.f[0];
}

// ---------------- GEMM: C[M,N] = act(A[M,K] x W[N,K]^T + bias) ----------------
// 256 thr = 8 waves; tile 64(M) x 64(N); K chunks of 32; register double-buffered
// staging so next chunk's global loads overlap the WMMAs. M%64==0, K%32==0.
__global__ __launch_bounds__(256) void k_gemm(
    const float* __restrict__ A, const float* __restrict__ W,
    const float* __restrict__ bias, float* __restrict__ C,
    int M, int N, int K, int act) {
  __shared__ __align__(16) _Float16 sA[64 * 40];
  __shared__ __align__(16) _Float16 sW[64 * 40];
  const int tid = threadIdx.x;
  const int wave = tid >> 5;
  const int m0 = blockIdx.x * 64, n0 = blockIdx.y * 64;
  const int mi = (wave & 3) * 16, nb = (wave >> 2) * 32;
  const int sr = tid >> 2, sc = (tid & 3) * 8;   // staging: one 8-seg per thread
  const float* aptr = A + (size_t)(m0 + sr) * K + sc;
  const float* wptr = W + (size_t)(n0 + sr) * K + sc;
  const bool wok = (n0 + sr < N);
  float4 fa0 = *(const float4*)(aptr);
  float4 fa1 = *(const float4*)(aptr + 4);
  float4 fw0 = make_float4(0.f, 0.f, 0.f, 0.f), fw1 = fw0;
  if (wok) { fw0 = *(const float4*)(wptr); fw1 = *(const float4*)(wptr + 4); }
  CFrag a0, a1;
  for (int r = 0; r < 8; ++r) { a0.f[r] = 0.f; a1.f[r] = 0.f; }
  for (int k0 = 0; k0 < K; k0 += 32) {
    *(uint4*)&sA[sr * 40 + sc] = cvt8(fa0, fa1, 1.f);
    *(uint4*)&sW[sr * 40 + sc] = cvt8(fw0, fw1, 1.f);
    __syncthreads();
    if (k0 + 32 < K) {  // issue next chunk's loads; wait lands next iteration
      aptr += 32; wptr += 32;
      fa0 = *(const float4*)(aptr);
      fa1 = *(const float4*)(aptr + 4);
      if (wok) { fw0 = *(const float4*)(wptr); fw1 = *(const float4*)(wptr + 4); }
    }
    v16h a = ld_fragA(sA + mi * 40, 40);
    a0.v = wmma32(a, ld_fragB(sW + nb * 40, 40), a0.v);
    a1.v = wmma32(a, ld_fragB(sW + (nb + 16) * 40, 40), a1.v);
    __syncthreads();
  }
  const int lane = tid & 31;
  const int roff = (lane >> 4) << 3, ncol = lane & 15;
  for (int r = 0; r < 8; ++r) {
    int m = m0 + mi + r + roff;
    int n1 = n0 + nb + ncol, n2 = n1 + 16;
    if (n1 < N) {
      float v = a0.f[r] + (bias ? bias[n1] : 0.f);
      if (act == 1) v = fmaxf(v, 0.f);
      C[(size_t)m * N + n1] = v;
    }
    if (n2 < N) {
      float v = a1.f[r] + (bias ? bias[n2] : 0.f);
      if (act == 1) v = fmaxf(v, 0.f);
      C[(size_t)m * N + n2] = v;
    }
  }
}

// ---------------- conv1d, channels-last: Y[b,t,co], 'same' padding ----------------
// X:[B][T][Cin], Wt:[Cout][Cin][KS]. grid=(B, T/64, Cout/64). Cin,Cout %32==0.
template <int KS, int ACT>
__global__ __launch_bounds__(256) void k_conv(
    const float* __restrict__ X, const float* __restrict__ Wt,
    const float* __restrict__ bias, float* __restrict__ Y,
    int T_, int Cin, int Cout) {
  constexpr int PAD = KS / 2;
  constexpr int EXT = 64 + KS - 1;
  constexpr int ROWLEN = 32 * KS;
  __shared__ __align__(16) _Float16 sA[EXT * 40];
  __shared__ __align__(16) _Float16 sW[KS * 64 * 40];  // all taps staged at once
  const int tid = threadIdx.x;
  const int wave = tid >> 5;
  const int b = blockIdx.x;
  const int t0 = blockIdx.y * 64;
  const int co0 = blockIdx.z * 64;
  const int mi = (wave & 3) * 16, nb = (wave >> 2) * 32;
  CFrag a0, a1;
  for (int r = 0; r < 8; ++r) { a0.f[r] = 0.f; a1.f[r] = 0.f; }
  for (int ci0 = 0; ci0 < Cin; ci0 += 32) {
    __syncthreads();
    // stage activations (EXT x 32) as 8-wide segments; zero outside [0,T)
    for (int s = tid; s < EXT * 4; s += 256) {
      int r = s >> 2, c = (s & 3) * 8;
      int t = t0 + r - PAD;
      uint4 u;
      if (t >= 0 && t < T_) {
        const float* src = X + ((size_t)(b * T_ + t)) * Cin + ci0 + c;
        float4 f0 = *(const float4*)(src);
        float4 f1 = *(const float4*)(src + 4);
        u = cvt8(f0, f1, 1.f);
        if (ci0 + 32 < Cin) __builtin_prefetch(src + 32, 0, 0);  // next Cin chunk
      } else {
        u.x = 0u; u.y = 0u; u.z = 0u; u.w = 0u;
      }
      *(uint4*)&sA[r * 40 + c] = u;
    }
    // stage weights: 64 rows of (32 ci * KS) contiguous floats
    for (int i = tid; i < 64 * ROWLEN; i += 256) {
      int n = i / ROWLEN, j = i - n * ROWLEN;
      int ci = j / KS, dt = j - ci * KS;
      const float* wsrc = Wt + ((size_t)(co0 + n) * Cin + ci0) * KS + j;
      sW[(dt * 64 + n) * 40 + ci] = cvt1(*wsrc);
      if (ci0 + 32 < Cin) __builtin_prefetch(wsrc + ROWLEN, 0, 0);
    }
    __syncthreads();
#pragma unroll
    for (int dt = 0; dt < KS; ++dt) {
      v16h a = ld_fragA(sA + (mi + dt) * 40, 40);
      const _Float16* wb = sW + (dt * 64 + nb) * 40;
      a0.v = wmma32(a, ld_fragB(wb, 40), a0.v);
      a1.v = wmma32(a, ld_fragB(wb + 16 * 40, 40), a1.v);
    }
  }
  const int lane = tid & 31;
  const int roff = (lane >> 4) << 3, ncol = lane & 15;
  for (int r = 0; r < 8; ++r) {
    int t = t0 + mi + r + roff;
    int n1 = co0 + nb + ncol, n2 = n1 + 16;
    float v1 = a0.f[r] + bias[n1];
    float v2 = a1.f[r] + bias[n2];
    if (ACT == 1) { v1 = fmaxf(v1, 0.f); v2 = fmaxf(v2, 0.f); }
    Y[((size_t)(b * T_ + t)) * Cout + n1] = v1;
    Y[((size_t)(b * T_ + t)) * Cout + n2] = v2;
  }
}

// ---------------- flash attention (NH=2, DH=128), one wave per (b,head,16 q) ------
__global__ __launch_bounds__(32) void k_attn(
    const float* __restrict__ qkv, const int* __restrict__ klen,
    float* __restrict__ O, int T_) {
  __shared__ __align__(16) _Float16 sQ[16 * 136];
  __shared__ __align__(16) _Float16 sK[32 * 136];
  __shared__ __align__(16) _Float16 sV[128 * 40];
  __shared__ __align__(16) _Float16 sP[16 * 40];
  const int b = blockIdx.x >> 1, head = blockIdx.x & 1;
  const int q0 = blockIdx.y * 16;
  const int lane = threadIdx.x & 31;
  const int len = klen[b];
  const float qscale = 0.08838834764831845f;  // 1/sqrt(128)
#pragma unroll
  for (int s = lane; s < 16 * 16; s += 32) {  // Q: 16 rows x 16 8-wide segs
    int r = s >> 4, c = (s & 15) * 8;
    const float* src = qkv + ((size_t)(b * T_ + q0 + r)) * 768 + head * 128 + c;
    float4 f0 = *(const float4*)(src);
    float4 f1 = *(const float4*)(src + 4);
    *(uint4*)&sQ[r * 136 + c] = cvt8(f0, f1, qscale);
  }
  __syncthreads();
  v16h qf[4];
#pragma unroll
  for (int kc = 0; kc < 4; ++kc) qf[kc] = ld_fragA(sQ + kc * 32, 136);
  CFrag o[8];
  float rm[8], rs[8];
  for (int c = 0; c < 8; ++c)
    for (int r = 0; r < 8; ++r) o[c].f[r] = 0.f;
  for (int r = 0; r < 8; ++r) { rm[r] = -3.0e38f; rs[r] = 0.f; }
  const int ncol = lane & 15;
  const int roff = (lane >> 4) << 3;
  for (int j0 = 0; j0 < T_; j0 += 32) {
#pragma unroll
    for (int s = lane; s < 32 * 16; s += 32) {  // K: 32 rows x 16 8-wide segs
      int r = s >> 4, c = (s & 15) * 8;
      const float* src =
          qkv + ((size_t)(b * T_ + j0 + r)) * 768 + 256 + head * 128 + c;
      float4 f0 = *(const float4*)(src);
      float4 f1 = *(const float4*)(src + 4);
      *(uint4*)&sK[r * 136 + c] = cvt8(f0, f1, 1.f);
      if (j0 + 32 < T_) __builtin_prefetch(src + 32 * 768, 0, 0);  // next key blk
    }
#pragma unroll
    for (int s = lane; s < 32 * 32; s += 32) {  // V transposed: vec loads, b16 st
      int kr = s >> 5, d4 = (s & 31) * 4;
      const float* src =
          qkv + ((size_t)(b * T_ + j0 + kr)) * 768 + 512 + head * 128 + d4;
      float4 f = *(const float4*)(src);
      P2 pa, pb;
      pa.h = __builtin_amdgcn_cvt_pkrtz(f.x, f.y);
      pb.h = __builtin_amdgcn_cvt_pkrtz(f.z, f.w);
      sV[(d4 + 0) * 40 + kr] = pa.f[0];
      sV[(d4 + 1) * 40 + kr] = pa.f[1];
      sV[(d4 + 2) * 40 + kr] = pb.f[0];
      sV[(d4 + 3) * 40 + kr] = pb.f[1];
    }
    __syncthreads();
    CFrag s0, s1;
    for (int r = 0; r < 8; ++r) { s0.f[r] = 0.f; s1.f[r] = 0.f; }
#pragma unroll
    for (int kc = 0; kc < 4; ++kc) {
      s0.v = wmma32(qf[kc], ld_fragB(sK + kc * 32, 136), s0.v);
      s1.v = wmma32(qf[kc], ld_fragB(sK + 16 * 136 + kc * 32, 136), s1.v);
    }
    if (j0 + ncol >= len)      for (int r = 0; r < 8; ++r) s0.f[r] = -1e9f;
    if (j0 + 16 + ncol >= len) for (int r = 0; r < 8; ++r) s1.f[r] = -1e9f;
#pragma unroll
    for (int r = 0; r < 8; ++r) {  // online softmax; rows live in half-waves
      float mx = fmaxf(s0.f[r], s1.f[r]);
      for (int d = 8; d >= 1; d >>= 1) mx = fmaxf(mx, __shfl_xor(mx, d));
      float nm = fmaxf(rm[r], mx);
      float corr = __expf(rm[r] - nm);
      float p0 = __expf(s0.f[r] - nm), p1 = __expf(s1.f[r] - nm);
      float sum = p0 + p1;
      for (int d = 8; d >= 1; d >>= 1) sum += __shfl_xor(sum, d);
      rs[r] = rs[r] * corr + sum;
      rm[r] = nm;
      s0.f[r] = p0; s1.f[r] = p1;
      for (int c = 0; c < 8; ++c) o[c].f[r] *= corr;
    }
    __syncthreads();
#pragma unroll
    for (int r = 0; r < 8; ++r) {  // C-layout -> A-layout via LDS
      sP[(r + roff) * 40 + ncol]      = cvt1(s0.f[r]);
      sP[(r + roff) * 40 + 16 + ncol] = cvt1(s1.f[r]);
    }
    __syncthreads();
    v16h pf = ld_fragA(sP, 40);
#pragma unroll
    for (int c = 0; c < 8; ++c)
      o[c].v = wmma32(pf, ld_fragB(sV + (c * 16) * 40, 40), o[c].v);
    __syncthreads();
  }
  for (int c = 0; c < 8; ++c)
    for (int r = 0; r < 8; ++r) {
      int t = q0 + r + roff;
      O[((size_t)(b * T_ + t)) * 256 + head * 128 + c * 16 + ncol] = o[c].f[r] / rs[r];
    }
}

// ---------------- residual + LayerNorm over H (block=256 per row) ----------------
__global__ __launch_bounds__(256) void k_addln(
    const float* __restrict__ X, const float* __restrict__ R,
    const float* __restrict__ g, const float* __restrict__ be,
    float* __restrict__ Y, int H_) {
  __shared__ float red[256];
  const int row = blockIdx.x, tid = threadIdx.x;
  float v = 0.f;
  if (tid < H_) {
    v = X[(size_t)row * H_ + tid];
    if (R) v += R[(size_t)row * H_ + tid];
  }
  red[tid] = v;
  __syncthreads();
  for (int s = 128; s > 0; s >>= 1) { if (tid < s) red[tid] += red[tid + s]; __syncthreads(); }
  float mean = red[0] / H_;
  __syncthreads();
  float d = v - mean;
  red[tid] = (tid < H_) ? d * d : 0.f;
  __syncthreads();
  for (int s = 128; s > 0; s >>= 1) { if (tid < s) red[tid] += red[tid + s]; __syncthreads(); }
  float var = red[0] / H_;
  if (tid < H_) Y[(size_t)row * H_ + tid] = d * rsqrtf(var + 1e-5f) * g[tid] + be[tid];
}

// ---------------- row-dot + sigmoid ----------------
__global__ __launch_bounds__(256) void k_rowdot(
    const float* __restrict__ X, const float* __restrict__ w,
    const float* __restrict__ b0, float* __restrict__ out, int H_) {
  __shared__ float red[256];
  const int row = blockIdx.x, tid = threadIdx.x;
  red[tid] = (tid < H_) ? X[(size_t)row * H_ + tid] * w[tid] : 0.f;
  __syncthreads();
  for (int s = 128; s > 0; s >>= 1) { if (tid < s) red[tid] += red[tid + s]; __syncthreads(); }
  if (tid == 0) out[row] = 1.f / (1.f + __expf(-(red[0] + b0[0])));
}

// ---------------- sinusoidal PE table (module constant in reference) --------------
__global__ void k_pe(float* __restrict__ PEb) {
  const int pos = blockIdx.x, d = threadIdx.x;
  float div = __expf(-(float)(d & ~1) * (9.210340371976184f / 256.0f));  // ln(1e4)
  float a = (float)pos * div;
  PEb[pos * 256 + d] = (d & 1) ? __cosf(a) : __sinf(a);
}

__global__ void k_embed(const int* __restrict__ ids, const float* __restrict__ emb,
                        const float* __restrict__ PEb, float* __restrict__ X) {
  const int row = blockIdx.x, tid = threadIdx.x;  // row = b*T + t
  const int t = row & 255;
  const int id = ids[row];
  X[(size_t)row * 256 + tid] = emb[(size_t)id * 256 + tid] + PEb[t * 256 + tid];
}

__global__ void k_durcum(const int* __restrict__ dur, const float* __restrict__ bneed,
                         int* __restrict__ cum, int* __restrict__ mlen) {
  const int b = threadIdx.x;
  if (b < BZ) {
    int c = 0;
    for (int t = 0; t < TT; ++t) {
      c += dur[b * TT + t] + ((bneed[b * TT + t] > 0.7f) ? 2 : 0);
      cum[b * TT + t] = c;
    }
    mlen[b] = c;
  }
}

__device__ inline int ss_left(float v, float lo, float hi, int n) {
  float step = (hi - lo) / (float)(n - 1);
  int i = (v <= lo) ? 0 : (int)ceilf((v - lo) / step);
  if (i > n - 1) i = n - 1;
  if (i < 0) i = 0;
  return i;
}

__global__ void k_adapt(const float* __restrict__ enc, const float* __restrict__ pitch,
                        const float* __restrict__ energy, const float* __restrict__ preds,
                        const float* __restrict__ pe_emb, const float* __restrict__ brn,
                        float* __restrict__ out) {
  const int row = blockIdx.x, tid = threadIdx.x;
  float pv = fminf(fmaxf(pitch[row], -3.f), 3.f);
  float ev = fminf(fmaxf(energy[row], -3.f), 3.f);
  float bv = preds[row], rv = preds[BT + row], nv = preds[2 * BT + row];
  int pi = ss_left(pv, -3.f, 3.f, 256);
  int ei = ss_left(ev, -3.f, 3.f, 256);
  int bi = ss_left(bv, 0.f, 1.f, 64);
  int ri = ss_left(rv, 0.f, 1.f, 64);
  int ni = ss_left(nv, 0.f, 1.f, 64);
  out[(size_t)row * 256 + tid] = enc[(size_t)row * 256 + tid]
      + pe_emb[pi * 256 + tid] + pe_emb[65536 + ei * 256 + tid]
      + brn[bi * 256 + tid] + brn[16384 + ri * 256 + tid] + brn[32768 + ni * 256 + tid];
}

__global__ void k_expand(const float* __restrict__ xa, const int* __restrict__ cum,
                         const int* __restrict__ mlen, const float* __restrict__ PEb,
                         float* __restrict__ xd) {
  const int idx = blockIdx.x;  // b*MEL + f
  const int b = idx / MELX, f = idx - b * MELX;
  const int tid = threadIdx.x;
  const int* c = cum + b * TT;
  int lo = 0, hi = TT;  // searchsorted(cum, f, 'right')
  while (lo < hi) { int mid = (lo + hi) >> 1; if (c[mid] <= f) lo = mid + 1; else hi = mid; }
  int tok = (lo > TT - 1) ? (TT - 1) : lo;
  float v = (f < mlen[b]) ? xa[((size_t)(b * TT + tok)) * 256 + tid] : 0.f;
  xd[(size_t)idx * 256 + tid] = v + PEb[f * 256 + tid];
}

__global__ void k_gate(const float* __restrict__ a, const float* __restrict__ g,
                       float* __restrict__ out, int n) {
  const int i = blockIdx.x * 256 + threadIdx.x;
  if (i < n) out[i] = a[i] * (1.f / (1.f + __expf(-g[i])));
}

// ============================ host orchestration ============================
extern "C" void kernel_launch(void* const* d_in, const int* in_sizes, int n_in,
                              void* d_out, int out_size, void* d_ws, size_t ws_size,
                              hipStream_t stream) {
  (void)in_sizes; (void)n_in; (void)out_size; (void)ws_size;
  const int*   ids  = (const int*)d_in[0];
  const int*   lens = (const int*)d_in[1];
  const int*   tdur = (const int*)d_in[2];
  const float* tpit = (const float*)d_in[3];
  const float* tene = (const float*)d_in[4];
  const float* emb  = (const float*)d_in[5];
  const float* enc_w[12]; for (int i = 0; i < 12; ++i) enc_w[i] = (const float*)d_in[6 + i];
  const float* dec_w[12]; for (int i = 0; i < 12; ++i) dec_w[i] = (const float*)d_in[18 + i];
  const float* vp_cw  = (const float*)d_in[30];
  const float* vp_cb  = (const float*)d_in[31];
  const float* vp_lns = (const float*)d_in[32];
  const float* vp_lnb = (const float*)d_in[33];
  const float* vp_pw  = (const float*)d_in[34];
  const float* vp_pb  = (const float*)d_in[35];
  const float* pe_emb = (const float*)d_in[36];
  const float* brn    = (const float*)d_in[37];
  const float* bg_w1  = (const float*)d_in[38];
  const float* bg_b1  = (const float*)d_in[39];
  const float* bg_w2  = (const float*)d_in[40];
  const float* bg_b2  = (const float*)d_in[41];
  const float* mel_w  = (const float*)d_in[42];
  const float* mel_b  = (const float*)d_in[43];
  const float* spec_w = (const float*)d_in[44];
  const float* spec_b = (const float*)d_in[45];

  char* ws = (char*)d_ws;
  size_t off = 0;
  auto alloc = [&](size_t bytes) -> void* {
    void* p = ws + off;
    off += (bytes + 255) & ~(size_t)255;
    return p;
  };
  float* PEb   = (float*)alloc((size_t)MELX * 256 * 4);
  float* xe    = (float*)alloc((size_t)BT * 256 * 4);
  float* xd    = (float*)alloc((size_t)BZ * MELX * 256 * 4);
  float* big   = (float*)alloc((size_t)BZ * MELX * 1024 * 4);  // qkv & conv-hidden
  float* ta    = (float*)alloc((size_t)BZ * MELX * 256 * 4);
  float* tb    = (float*)alloc((size_t)BZ * MELX * 256 * 4);
  float* preds = (float*)alloc((size_t)3 * BT * 4);
  float* bneed = (float*)alloc((size_t)BT * 4);
  int*   cum   = (int*)alloc((size_t)BT * 4);
  int*   mlen  = (int*)alloc((size_t)BZ * 4);
  float* xa    = (float*)alloc((size_t)BT * 256 * 4);
  float* m1    = (float*)alloc((size_t)BZ * MELX * NMEL * 4);
  float* m2    = (float*)alloc((size_t)BZ * MELX * NMEL * 4);

  k_pe<<<MELX, 256, 0, stream>>>(PEb);
  k_embed<<<BT, 256, 0, stream>>>(ids, emb, PEb, xe);

  auto fft = [&](float* x, int Tlen, const int* klen, const float* const* w, int l) {
    const int Mr = BZ * Tlen;
    const float* qkvW = w[0] + (size_t)l * 768 * 256;
    const float* qkvB = w[1] + l * 768;
    const float* oW   = w[2] + (size_t)l * 256 * 256;
    const float* oB   = w[3] + l * 256;
    const float* n1s  = w[4] + l * 256;
    const float* n1b  = w[5] + l * 256;
    const float* c1w  = w[6] + (size_t)l * 1024 * 256 * 9;
    const float* c1b  = w[7] + l * 1024;
    const float* c2w  = w[8] + (size_t)l * 256 * 1024 * 9;
    const float* c2b  = w[9] + l * 256;
    const float* n2s  = w[10] + l * 256;
    const float* n2b  = w[11] + l * 256;
    k_gemm<<<dim3(Mr / 64, 12), 256, 0, stream>>>(x, qkvW, qkvB, big, Mr, 768, 256, 0);
    k_attn<<<dim3(BZ * 2, Tlen / 16), 32, 0, stream>>>(big, klen, ta, Tlen);
    k_gemm<<<dim3(Mr / 64, 4), 256, 0, stream>>>(ta, oW, oB, tb, Mr, 256, 256, 0);
    k_addln<<<Mr, 256, 0, stream>>>(x, tb, n1s, n1b, x, 256);
    k_conv<9, 1><<<dim3(BZ, Tlen / 64, 16), 256, 0, stream>>>(x, c1w, c1b, big,
                                                              Tlen, 256, 1024);
    k_conv<9, 0><<<dim3(BZ, Tlen / 64, 4), 256, 0, stream>>>(big, c2w, c2b, tb,
                                                             Tlen, 1024, 256);
    k_addln<<<Mr, 256, 0, stream>>>(x, tb, n2s, n2b, x, 256);
  };

  for (int l = 0; l < 4; ++l) fft(xe, TT, lens, enc_w, l);

  // variance predictors 3,4,5 (breath/rough/nasal); 0..2 are dead in the reference
  for (int i = 3; i < 6; ++i) {
    const float* src = xe;
    for (int j = 0; j < 2; ++j) {
      const float* cw = vp_cw + ((size_t)(i * 2 + j)) * 256 * 256 * 3;
      const float* cb = vp_cb + (i * 2 + j) * 256;
      const float* ls = vp_lns + (i * 2 + j) * 256;
      const float* lb = vp_lnb + (i * 2 + j) * 256;
      k_conv<3, 1><<<dim3(BZ, 4, 4), 256, 0, stream>>>(src, cw, cb, ta, TT, 256, 256);
      k_addln<<<BT, 256, 0, stream>>>(ta, nullptr, ls, lb, tb, 256);
      src = tb;
    }
    k_rowdot<<<BT, 256, 0, stream>>>(tb, vp_pw + i * 256, vp_pb + i,
                                     preds + (i - 3) * BT, 256);
  }

  // breath gate
  k_gemm<<<dim3(BT / 64, 2), 256, 0, stream>>>(xe, bg_w1, bg_b1, ta, BT, 128, 256, 1);
  k_rowdot<<<BT, 256, 0, stream>>>(ta, bg_w2, bg_b2, bneed, 128);
  k_durcum<<<1, 32, 0, stream>>>(tdur, bneed, cum, mlen);
  k_adapt<<<BT, 256, 0, stream>>>(xe, tpit, tene, preds, pe_emb, brn, xa);
  k_expand<<<BZ * MELX, 256, 0, stream>>>(xa, cum, mlen, PEb, xd);

  for (int l = 0; l < 4; ++l) fft(xd, MELX, mlen, dec_w, l);

  const int Mr = BZ * MELX;
  k_gemm<<<dim3(Mr / 64, 2), 256, 0, stream>>>(xd, mel_w, mel_b, m1, Mr, NMEL, 256, 0);
  k_gemm<<<dim3(Mr / 64, 2), 256, 0, stream>>>(xd, spec_w, spec_b, m2, Mr, NMEL, 256, 0);
  const int n = Mr * NMEL;
  k_gate<<<(n + 255) / 256, 256, 0, stream>>>(m1, m2, (float*)d_out, n);
}
